// InterpretableMultiHeadAttention_1709396984558
// MI455X (gfx1250) — matrix operand order repaired and checked by reading, not compiled, not run on hardware
//
#include <hip/hip_runtime.h>
#include <hip/hip_bf16.h>

typedef __attribute__((ext_vector_type(16))) __bf16 v16bf;
typedef __attribute__((ext_vector_type(8)))  float  v8f;

static constexpr int Bq = 2, Sq = 2048, Dq = 1024, Hq = 8, DKq = 128;

// ---------------------------------------------------------------------------
// WMMA fragment helpers (bf16 16x16x32, wave32).
//
// A (16x32, MxK): lane<16 -> row M=lane,     K = {k0+0..7, k0+16..23}
//                 lane>=16 -> row M=lane-16, K = {k0+8..15, k0+24..31}
// B (32x16, KxN): lane gives N = n0 + (lane&15);
//                 lane<16 -> K = {k0+0..7, k0+16..23}, lane>=16 -> K = {k0+8..15, k0+24..31}
// C/D (16x16 f32): elem i -> M = 8*(lane>>4) + i, N = lane&15
// ---------------------------------------------------------------------------

__device__ __forceinline__ v16bf load_a_bf16(const __bf16* base, int ld, int row0, int k0) {
  const int lane = threadIdx.x & 31;
  const int row  = row0 + (lane & 15);
  const int kb   = k0 + ((lane >> 4) << 3);
  const __bf16* p = base + (size_t)row * ld + kb;
  union { uint4 u; __bf16 h[8]; } lo, hi;
  lo.u = *(const uint4*)(p);
  hi.u = *(const uint4*)(p + 16);
  v16bf a;
#pragma unroll
  for (int j = 0; j < 8; ++j) { a[j] = lo.h[j]; a[j + 8] = hi.h[j]; }
  return a;
}

// B operand stored "column major": element (k, n) at base[n*ld + k]
__device__ __forceinline__ v16bf load_b_bf16(const __bf16* base, int ld, int n0, int k0) {
  const int lane = threadIdx.x & 31;
  const int n  = n0 + (lane & 15);
  const int kb = k0 + ((lane >> 4) << 3);
  const __bf16* p = base + (size_t)n * ld + kb;
  union { uint4 u; __bf16 h[8]; } lo, hi;
  lo.u = *(const uint4*)(p);
  hi.u = *(const uint4*)(p + 16);
  v16bf b;
#pragma unroll
  for (int j = 0; j < 8; ++j) { b[j] = lo.h[j]; b[j + 8] = hi.h[j]; }
  return b;
}

// A fragment gathered from an f32 LDS tile (rows 0..15), converted to bf16
__device__ __forceinline__ v16bf load_a_lds_f32(const float* sm, int ld, int k0) {
  const int lane = threadIdx.x & 31;
  const int row  = lane & 15;
  const int kb   = k0 + ((lane >> 4) << 3);
  const float* p = sm + row * ld + kb;
  v16bf a;
#pragma unroll
  for (int j = 0; j < 8; ++j) { a[j] = (__bf16)p[j]; a[j + 8] = (__bf16)p[16 + j]; }
  return a;
}

__device__ __forceinline__ v8f wmma_bf16(v16bf a, v16bf b, v8f c) {
  return __builtin_amdgcn_wmma_f32_16x16x32_bf16(false, a, false, b, (short)0, c, false, false);
}

// ---------------------------------------------------------------------------
// Prep kernels
// ---------------------------------------------------------------------------

__global__ void k_cvt_bf16(const float* __restrict__ in, __bf16* __restrict__ out, long n) {
  for (long i = (long)blockIdx.x * blockDim.x + threadIdx.x; i < n;
       i += (long)gridDim.x * blockDim.x)
    out[i] = (__bf16)in[i];
}

// in: [batch, M, N] f32 -> out: [batch, N, M] bf16
__global__ void k_transpose_bf16(const float* __restrict__ in, __bf16* __restrict__ out,
                                 int M, int N, int batch) {
  const long mn = (long)M * N, total = mn * batch;
  for (long i = (long)blockIdx.x * blockDim.x + threadIdx.x; i < total;
       i += (long)gridDim.x * blockDim.x) {
    long b = i / mn;
    long r = i - b * mn;
    int m = (int)(r / N);
    int n = (int)(r - (long)m * N);
    out[b * mn + (long)n * M + m] = (__bf16)in[i];
  }
}

// ---------------------------------------------------------------------------
// Per-head projection: out[b,h,s,dk] = x[b,s,:] @ W[h,:,dk] + bias[h,dk]
// grid = B*H*(S/16), block = 256 (8 waves; wave w -> dk tile w)
// ---------------------------------------------------------------------------
__global__ void __launch_bounds__(256) k_proj_head(
    const __bf16* __restrict__ x,    // [B,S,D] bf16
    const __bf16* __restrict__ wT,   // [H,DK,D] bf16 (K contiguous)
    const float*  __restrict__ bias, // [H,DK]
    __bf16*       __restrict__ out)  // [B,H,S,DK] bf16
{
  const int tiles = Sq / 16;
  int t = blockIdx.x;
  const int st = t % tiles; t /= tiles;
  const int h = t % Hq;
  const int b = t / Hq;
  const int row0 = st * 16;
  const __bf16* xb = x  + (size_t)b * Sq * Dq;
  const __bf16* wh = wT + (size_t)h * DKq * Dq;
  const int w = threadIdx.x >> 5, lane = threadIdx.x & 31;
  const int n0 = w * 16;
  v8f acc = {};
#pragma unroll 4
  for (int k0 = 0; k0 < Dq; k0 += 32) {
    v16bf a  = load_a_bf16(xb, Dq, row0, k0);
    v16bf bb = load_b_bf16(wh, Dq, n0, k0);
    acc = wmma_bf16(a, bb, acc);
  }
  const int n = n0 + (lane & 15);
  const float bv = bias[h * DKq + n];
  __bf16* op = out + ((size_t)b * Hq + h) * Sq * DKq;
#pragma unroll
  for (int i = 0; i < 8; ++i) {
    const int m = row0 + ((lane >> 4) << 3) + i;
    op[(size_t)m * DKq + n] = (__bf16)(acc[i] + bv);
  }
}

// Shared V projection, stored TRANSPOSED: outT[b,dk,s] (= B operand of probs@V)
// grid = B*(S/16), block = 256
__global__ void __launch_bounds__(256) k_proj_v(
    const __bf16* __restrict__ x,    // [B,S,D] bf16
    const __bf16* __restrict__ wT,   // [DK,D] bf16
    const float*  __restrict__ bias, // [DK]
    __bf16*       __restrict__ outT) // [B,DK,S] bf16
{
  const int tiles = Sq / 16;
  const int st = blockIdx.x % tiles;
  const int b  = blockIdx.x / tiles;
  const int row0 = st * 16;
  const __bf16* xb = x + (size_t)b * Sq * Dq;
  const int w = threadIdx.x >> 5, lane = threadIdx.x & 31;
  const int n0 = w * 16;
  v8f acc = {};
#pragma unroll 4
  for (int k0 = 0; k0 < Dq; k0 += 32) {
    v16bf a  = load_a_bf16(xb, Dq, row0, k0);
    v16bf bb = load_b_bf16(wT, Dq, n0, k0);
    acc = wmma_bf16(a, bb, acc);
  }
  const int n = n0 + (lane & 15);
  const float bv = bias[n];
#pragma unroll
  for (int i = 0; i < 8; ++i) {
    const int m = row0 + ((lane >> 4) << 3) + i;
    outT[((size_t)b * DKq + n) * Sq + m] = (__bf16)(acc[i] + bv);
  }
}

// ---------------------------------------------------------------------------
// Fused attention stripe: one workgroup = 16 query rows x full key length.
// 1) scores (WMMA)  -> LDS f32 [16][S]   (128 KB of the 320 KB WGP LDS)
// 2) wave-parallel softmax (2 rows per wave, shuffle reductions, no barriers)
//    + single non-temporal streaming write of probs to the attn output
// 3) head = probs @ V  (A from LDS w/ on-the-fly bf16 cvt; B = vsT)
// grid = B*H*(S/16), block = 256
// ---------------------------------------------------------------------------
__global__ void __launch_bounds__(256) k_attn(
    const __bf16* __restrict__ qs,   // [B,H,S,DK]
    const __bf16* __restrict__ kpr,  // [B,H,S,DK]
    const __bf16* __restrict__ vsT,  // [B,DK,S]
    float*        __restrict__ head, // [B,H,S,DK]
    float*        __restrict__ attn) // [B,S,H,S]
{
  extern __shared__ float smem[];
  float* sc = smem; // 16 * S
  const int tiles = Sq / 16;
  int t = blockIdx.x;
  const int st = t % tiles; t /= tiles;
  const int h = t % Hq;
  const int b = t / Hq;
  const int row0 = st * 16;
  const int w = threadIdx.x >> 5, lane = threadIdx.x & 31;
  const __bf16* qb = qs  + ((size_t)b * Hq + h) * Sq * DKq;
  const __bf16* kb = kpr + ((size_t)b * Hq + h) * Sq * DKq;
  const float scale = 0.08838834764831845f; // 1/sqrt(128)

  // ---- scores: each wave owns 16 of the 128 column tiles ----
  v16bf aq[4];
#pragma unroll
  for (int kk = 0; kk < 4; ++kk) aq[kk] = load_a_bf16(qb, DKq, row0, kk * 32);
  for (int i = 0; i < 16; ++i) {
    const int ct = w * 16 + i;
    v8f acc = {};
#pragma unroll
    for (int kk = 0; kk < 4; ++kk) {
      v16bf bfrag = load_b_bf16(kb, DKq, ct * 16, kk * 32);
      acc = wmma_bf16(aq[kk], bfrag, acc);
    }
    const int n = ct * 16 + (lane & 15);
#pragma unroll
    for (int e = 0; e < 8; ++e) {
      const int m = ((lane >> 4) << 3) + e;
      sc[m * Sq + n] = acc[e] * scale;
    }
  }
  __syncthreads();

  // ---- softmax: wave w owns rows 2w, 2w+1; shuffle reductions, no barriers --
#pragma unroll
  for (int rr = 0; rr < 2; ++rr) {
    const int r = w * 2 + rr;
    float* row = sc + r * Sq;
    float mx = -3.402823466e38f;
    for (int c = lane; c < Sq; c += 32) mx = fmaxf(mx, row[c]);
#pragma unroll
    for (int m_ = 16; m_ > 0; m_ >>= 1) mx = fmaxf(mx, __shfl_xor(mx, m_, 32));
    float sum = 0.f;
    for (int c = lane; c < Sq; c += 32) {
      const float e = __expf(row[c] - mx);
      row[c] = e;
      sum += e;
    }
#pragma unroll
    for (int m_ = 16; m_ > 0; m_ >>= 1) sum += __shfl_xor(sum, m_, 32);
    const float inv = 1.f / sum;
    float* dst = attn + (((size_t)b * Sq + row0 + r) * Hq + h) * Sq;
    for (int c = lane; c < Sq; c += 32) {
      const float p = row[c] * inv;
      row[c] = p;
      __builtin_nontemporal_store(p, dst + c); // 268 MB stream: keep it out of L2
    }
  }
  __syncthreads();

  // ---- head = probs @ vs : wave w -> dk tile w ----
  const __bf16* vb = vsT + (size_t)b * DKq * Sq;
  const int n0 = w * 16;
  v8f acc2 = {};
  for (int kt = 0; kt < Sq / 32; ++kt) {
    v16bf a  = load_a_lds_f32(sc, Sq, kt * 32);
    v16bf bb = load_b_bf16(vb, Sq, n0, kt * 32);
    acc2 = wmma_bf16(a, bb, acc2);
  }
  float* hp = head + ((size_t)b * Hq + h) * Sq * DKq;
  const int n = n0 + (lane & 15);
#pragma unroll
  for (int e = 0; e < 8; ++e) {
    const int m = row0 + ((lane >> 4) << 3) + e;
    hp[(size_t)m * DKq + n] = acc2[e];
  }
}

// mean over heads -> bf16 A operand for the output projection
__global__ void k_head_mean(const float* __restrict__ head, __bf16* __restrict__ hm) {
  const long total = (long)Bq * Sq * DKq;
  for (long i = (long)blockIdx.x * blockDim.x + threadIdx.x; i < total;
       i += (long)gridDim.x * blockDim.x) {
    const int k = (int)(i % DKq);
    const long r = i / DKq;
    const int s = (int)(r % Sq);
    const int b = (int)(r / Sq);
    float acc = 0.f;
#pragma unroll
    for (int h = 0; h < Hq; ++h)
      acc += head[(((size_t)b * Hq + h) * Sq + s) * DKq + k];
    hm[i] = (__bf16)(acc * 0.125f);
  }
}

// out[b,s,:] = head_mean[b,s,:] @ Wh     grid = B*(S/16), block = 256
__global__ void __launch_bounds__(256) k_out_proj(
    const __bf16* __restrict__ hm,  // [B,S,DK] bf16
    const __bf16* __restrict__ whT, // [D,DK] bf16
    float*        __restrict__ out) // [B,S,D] f32
{
  const int tiles = Sq / 16;
  const int st = blockIdx.x % tiles;
  const int b  = blockIdx.x / tiles;
  const int row0 = st * 16;
  const int w = threadIdx.x >> 5, lane = threadIdx.x & 31;
  const __bf16* hb = hm + (size_t)b * Sq * DKq;
  v16bf af[4];
#pragma unroll
  for (int kk = 0; kk < 4; ++kk) af[kk] = load_a_bf16(hb, DKq, row0, kk * 32);
  for (int t = 0; t < 8; ++t) {
    const int n0 = (w * 8 + t) * 16;
    v8f acc = {};
#pragma unroll
    for (int kk = 0; kk < 4; ++kk) {
      v16bf bb = load_b_bf16(whT, DKq, n0, kk * 32);
      acc = wmma_bf16(af[kk], bb, acc);
    }
    const int n = n0 + (lane & 15);
#pragma unroll
    for (int e = 0; e < 8; ++e) {
      const int m = row0 + ((lane >> 4) << 3) + e;
      out[((size_t)b * Sq + m) * Dq + n] = acc[e];
    }
  }
}

// ---------------------------------------------------------------------------

extern "C" void kernel_launch(void* const* d_in, const int* in_sizes, int n_in,
                              void* d_out, int out_size, void* d_ws, size_t ws_size,
                              hipStream_t stream) {
  (void)in_sizes; (void)n_in; (void)out_size; (void)ws_size;

  const float* q  = (const float*)d_in[0];
  const float* k  = (const float*)d_in[1];
  const float* v  = (const float*)d_in[2];
  const float* Wq = (const float*)d_in[3];
  const float* bq = (const float*)d_in[4];
  const float* Wk = (const float*)d_in[5];
  const float* bk = (const float*)d_in[6];
  const float* Wv = (const float*)d_in[7];
  const float* bv = (const float*)d_in[8];
  const float* Wh = (const float*)d_in[9];

  float* out_ptr  = (float*)d_out;                  // [B,S,D]
  float* attn_ptr = out_ptr + (size_t)Bq * Sq * Dq; // [B,S,H,S]

  char* ws = (char*)d_ws;
  size_t off = 0;
  auto alloc = [&](size_t bytes) -> void* {
    void* p = ws + off;
    off += (bytes + 255) & ~(size_t)255;
    return p;
  };

  __bf16* qb  = (__bf16*)alloc((size_t)Bq * Sq * Dq * 2);
  __bf16* kbf = (__bf16*)alloc((size_t)Bq * Sq * Dq * 2);
  __bf16* vb  = (__bf16*)alloc((size_t)Bq * Sq * Dq * 2);
  __bf16* wqT = (__bf16*)alloc((size_t)Hq * DKq * Dq * 2);
  __bf16* wkT = (__bf16*)alloc((size_t)Hq * DKq * Dq * 2);
  __bf16* wvT = (__bf16*)alloc((size_t)DKq * Dq * 2);
  __bf16* whT = (__bf16*)alloc((size_t)Dq * DKq * 2);
  __bf16* qsb = (__bf16*)alloc((size_t)Bq * Hq * Sq * DKq * 2);
  __bf16* ksb = (__bf16*)alloc((size_t)Bq * Hq * Sq * DKq * 2);
  __bf16* vsT = (__bf16*)alloc((size_t)Bq * DKq * Sq * 2);
  float*  head= (float*) alloc((size_t)Bq * Hq * Sq * DKq * 4);
  __bf16* hm  = (__bf16*)alloc((size_t)Bq * Sq * DKq * 2);

  const long nqkv = (long)Bq * Sq * Dq;
  k_cvt_bf16<<<1024, 256, 0, stream>>>(q, qb,  nqkv);
  k_cvt_bf16<<<1024, 256, 0, stream>>>(k, kbf, nqkv);
  k_cvt_bf16<<<1024, 256, 0, stream>>>(v, vb,  nqkv);
  k_transpose_bf16<<<1024, 256, 0, stream>>>(Wq, wqT, Dq, DKq, Hq);
  k_transpose_bf16<<<1024, 256, 0, stream>>>(Wk, wkT, Dq, DKq, Hq);
  k_transpose_bf16<<<256,  256, 0, stream>>>(Wv, wvT, Dq, DKq, 1);
  k_transpose_bf16<<<256,  256, 0, stream>>>(Wh, whT, DKq, Dq, 1);

  const dim3 blk(256);
  k_proj_head<<<Bq * Hq * (Sq / 16), blk, 0, stream>>>(qb,  wqT, bq, qsb);
  k_proj_head<<<Bq * Hq * (Sq / 16), blk, 0, stream>>>(kbf, wkT, bk, ksb);
  k_proj_v  <<<Bq * (Sq / 16),       blk, 0, stream>>>(vb,  wvT, bv, vsT);

  const size_t smem = (size_t)16 * Sq * sizeof(float);
  k_attn<<<Bq * Hq * (Sq / 16), blk, smem, stream>>>(qsb, ksb, vsT, head, attn_ptr);

  k_head_mean<<<1024, 256, 0, stream>>>(head, hm);
  k_out_proj<<<Bq * (Sq / 16), blk, 0, stream>>>(hm, whT, out_ptr);
}
